// GLS_network_joint_25589415149981
// MI455X (gfx1250) — matrix operand order repaired and checked by reading, not compiled, not run on hardware
//
#include <hip/hip_runtime.h>
#include <hip/hip_bf16.h>
#include <math.h>

// ---------------- problem constants ----------------
static constexpr int B_    = 32;
static constexpr int N_    = 18;
static constexpr int T_    = 96;
static constexpr int P_    = 16;
static constexpr int TP_   = 112;   // T + P
static constexpr int CIN_  = 16;    // C_GLU_IN
static constexpr int COUT_ = 32;    // C_GLU_OUT
static constexpr int FIN_  = 3072;  // COUT * T
static constexpr int H_    = 4;
static constexpr int FOUT_ = 1536;  // GAT_OUT * T
static constexpr int HFOUT_= 6144;  // H * FOUT
static constexpr int MROWS_= 576;   // B * N

typedef __attribute__((ext_vector_type(16))) __bf16 v16bf;
typedef __attribute__((ext_vector_type(8)))  float  v8f;

// ---------------- start embed + pad ----------------
// xin[b,c,n,t] = (t<96) ? data[b,n,t]*w_start[c] + b_start[c] : 0
__global__ __launch_bounds__(256) void start_kernel(
    const float* __restrict__ data, const float* __restrict__ wst,
    const float* __restrict__ bst, float* __restrict__ xin) {
  int bn = blockIdx.x; int n = bn % N_; int b = bn / N_;
  const float* dp = data + ((size_t)b * N_ + n) * T_;
  for (int e = threadIdx.x; e < CIN_ * TP_; e += 256) {
    int c = e / TP_, t = e % TP_;
    float v = (t < T_) ? dp[t] * wst[c] + bst[c] : 0.0f;
    xin[(((size_t)b * CIN_ + c) * N_ + n) * TP_ + t] = v;
  }
}

// ---------------- causal dilated conv (K=2, kernel-height 1) ----------------
// out[b,co,n,t] = bias[co] + sum_ci w[co,ci,0]*in[t-d] + w[co,ci,1]*in[t]
__global__ __launch_bounds__(256) void conv_kernel(
    const float* __restrict__ in, const float* __restrict__ w,
    const float* __restrict__ bias, float* __restrict__ out,
    int Cin, int dil) {
  int bn = blockIdx.x; int n = bn % N_; int b = bn / N_;
  __shared__ float sin_[COUT_ * TP_];     // max 32*112 floats
  __shared__ float sw[COUT_ * COUT_ * 2]; // max 32*32*2
  for (int e = threadIdx.x; e < Cin * TP_; e += 256)
    sin_[e] = in[(((size_t)b * Cin + (e / TP_)) * N_ + n) * TP_ + (e % TP_)];
  for (int e = threadIdx.x; e < COUT_ * Cin * 2; e += 256)
    sw[e] = w[e];
  __syncthreads();
  for (int e = threadIdx.x; e < COUT_ * TP_; e += 256) {
    int co = e / TP_, t = e % TP_;
    float acc = bias[co];
    for (int ci = 0; ci < Cin; ++ci) {
      float x1 = sin_[ci * TP_ + t];
      float x0 = (t >= dil) ? sin_[ci * TP_ + t - dil] : 0.0f;
      acc += sw[(co * Cin + ci) * 2 + 0] * x0 + sw[(co * Cin + ci) * 2 + 1] * x1;
    }
    out[(((size_t)b * COUT_ + co) * N_ + n) * TP_ + t] = acc;
  }
}

// ---------------- GLU gate: node -> split-bf16 A matrix, pos -> pel ----------------
__global__ __launch_bounds__(256) void gate_kernel(
    const float* __restrict__ F, const float* __restrict__ G,
    const float* __restrict__ R, const float* __restrict__ fc1w,
    const float* __restrict__ fc1b,
    __bf16* __restrict__ Ahi, __bf16* __restrict__ Alo,
    float* __restrict__ pel /* (N,B,1,P) in d_out */) {
  int bn = blockIdx.x; int n = bn % N_; int b = bn / N_;
  __shared__ float spos[COUT_ * P_];
  for (int e = threadIdx.x; e < COUT_ * TP_; e += 256) {
    int c = e / TP_, t = e % TP_;
    size_t idx = (((size_t)b * COUT_ + c) * N_ + n) * TP_ + t;
    float f = tanhf(F[idx]);
    float g = 1.0f / (1.0f + __expf(-G[idx]));
    float v = f * g + R[idx];
    v = v > 0.0f ? v : 0.0f;
    if (t < T_) {
      // nf flat layout: b*(C*N*T) + c*(N*T) + n*T + t == rows of (576 x 3072)
      size_t a = (size_t)b * (COUT_ * N_ * T_) + (size_t)c * (N_ * T_) + (size_t)n * T_ + t;
      __bf16 hi = (__bf16)v;
      Ahi[a] = hi;
      Alo[a] = (__bf16)(v - (float)hi);
    } else {
      spos[c * P_ + (t - T_)] = v;
    }
  }
  __syncthreads();
  if (threadIdx.x < P_) {
    int p = threadIdx.x;
    float s = fc1b[0];
    for (int c = 0; c < COUT_; ++c) s += fc1w[c] * spos[c * P_ + p];
    pel[((size_t)n * B_ + b) * P_ + p] = s;
  }
}

// ---------------- Kendall tau: S matrix ----------------
// feat[i,a] = pe[a,i] (pe viewed (18, 512)); S[a,b] = sum_{i,j} sgn(fa_i-fa_j)*sgn(fb_i-fb_j)
__global__ __launch_bounds__(256) void kendall_S_kernel(
    const float* __restrict__ pe, float* __restrict__ S) {
  int a = blockIdx.x / N_, bb = blockIdx.x % N_;
  __shared__ float fa[512], fb[512];
  for (int i = threadIdx.x; i < 512; i += 256) {
    fa[i] = pe[(size_t)a * 512 + i];
    fb[i] = pe[(size_t)bb * 512 + i];
  }
  __syncthreads();
  int acc = 0;
  for (int t = threadIdx.x; t < 512 * 512; t += 256) {
    int i = t >> 9, j = t & 511;
    float da = fa[i] - fa[j];
    float db = fb[i] - fb[j];
    int sa = (da > 0.0f) - (da < 0.0f);
    int sb = (db > 0.0f) - (db < 0.0f);
    acc += sa * sb;
  }
  __shared__ int red[256];
  red[threadIdx.x] = acc; __syncthreads();
  for (int off = 128; off > 0; off >>= 1) {
    if (threadIdx.x < off) red[threadIdx.x] += red[threadIdx.x + off];
    __syncthreads();
  }
  if (threadIdx.x == 0) S[blockIdx.x] = (float)red[0];
}

__global__ __launch_bounds__(384) void kendall_adj_kernel(
    const float* __restrict__ S, float* __restrict__ adj_ws,
    float* __restrict__ adj_out) {
  __shared__ float d[N_];
  if (threadIdx.x < N_) d[threadIdx.x] = S[threadIdx.x * N_ + threadIdx.x];
  __syncthreads();
  int t = threadIdx.x;
  if (t < N_ * N_) {
    const int g[N_] = {0,0,0,0,0,0,0, 1,1,1,1, 2,2, 3,3,3,3,3};
    int a = t / N_, b = t % N_;
    float tau = fabsf(S[t] / sqrtf(d[a] * d[b]));
    float thr = (g[a] == g[b]) ? 0.02f : 0.024f;
    float v = (tau > thr && a != b) ? 1.0f : 0.0f;
    adj_ws[t] = v;
    adj_out[t] = v;
  }
}

// ---------------- split-bf16 WMMA GEMM: C(576x6144) = A(576x3072) * W(3072x6144) + bias ----------------
// One wave -> one 16x16 C tile. 3 x v_wmma_f32_16x16x32_bf16 per K-step (hi*hi + lo*hi + hi*lo).
__global__ __launch_bounds__(256) void gemm_wmma_kernel(
    const __bf16* __restrict__ Ahi, const __bf16* __restrict__ Alo,
    const float* __restrict__ W, const float* __restrict__ bias,
    float* __restrict__ C) {
  const int lane = threadIdx.x & 31;
  const int wave = threadIdx.x >> 5;
  const int tile = blockIdx.x * 8 + wave;
  const int mt = tile / (HFOUT_ / 16);
  const int nt = tile % (HFOUT_ / 16);
  if (mt >= MROWS_ / 16) return; // wave-uniform; EXEC stays all-ones for WMMA
  const int m0 = mt * 16, n0 = nt * 16;
  const int half = lane >> 4;    // lanes 0-15 vs 16-31
  const int mr   = lane & 15;

  const __bf16* arh = Ahi + (size_t)(m0 + mr) * FIN_;
  const __bf16* arl = Alo + (size_t)(m0 + mr) * FIN_;
  const float*  wc  = W + (n0 + mr);

  v8f acc = {};
  for (int k0 = 0; k0 < FIN_; k0 += 32) {
    if (k0 + 32 < FIN_) __builtin_prefetch(wc + (size_t)(k0 + 32) * HFOUT_, 0, 0);
    v16bf ah, al, bh, bl;
    // A 16x32 bf16 layout: VGPR v<4: K = half*8 + 2v ; v>=4: K = 16 + half*8 + 2(v-4)
#pragma unroll
    for (int v = 0; v < 8; ++v) {
      int k = k0 + ((v & 4) << 2) + half * 8 + ((v & 3) << 1);
      ah[2 * v]     = arh[k];
      ah[2 * v + 1] = arh[k + 1];
      al[2 * v]     = arl[k];
      al[2 * v + 1] = arl[k + 1];
    }
    // B 32x16 bf16 layout: VGPR v<4: K = half*8 + 2v ; v>=4: K = 16 + half*8 + 2(v-4), N = lane&15
#pragma unroll
    for (int v = 0; v < 8; ++v) {
      int k = k0 + ((v & 4) << 2) + half * 8 + ((v & 3) << 1);
      float w0 = wc[(size_t)k * HFOUT_];
      float w1 = wc[(size_t)(k + 1) * HFOUT_];
      __bf16 h0 = (__bf16)w0, h1 = (__bf16)w1;
      bh[2 * v]     = h0;
      bh[2 * v + 1] = h1;
      bl[2 * v]     = (__bf16)(w0 - (float)h0);
      bl[2 * v + 1] = (__bf16)(w1 - (float)h1);
    }
    acc = __builtin_amdgcn_wmma_f32_16x16x32_bf16(false, ah, false, bh, (short)0, acc, false, false);
    acc = __builtin_amdgcn_wmma_f32_16x16x32_bf16(false, al, false, bh, (short)0, acc, false, false);
    acc = __builtin_amdgcn_wmma_f32_16x16x32_bf16(false, ah, false, bl, (short)0, acc, false, false);
  }
  // C/D layout: VGPR r: M = r + 8*half, N = lane&15
#pragma unroll
  for (int r = 0; r < 8; ++r) {
    int m = m0 + r + 8 * half;
    int n = n0 + mr;
    C[(size_t)m * HFOUT_ + n] = acc[r] + bias[n];
  }
}

// ---------------- attention scores ----------------
// scores[b,i,j,h] = sum_f leakyrelu(gl[b,i,h,f] + gr[b,j,h,f]) * att[h,f]
__global__ __launch_bounds__(256) void scores_kernel(
    const float* __restrict__ gl, const float* __restrict__ gr,
    const float* __restrict__ att, float* __restrict__ scores) {
  int bij = blockIdx.x;
  int j = bij % N_, i = (bij / N_) % N_, b = bij / (N_ * N_);
  const float* glp = gl + (size_t)(b * N_ + i) * HFOUT_;
  const float* grp = gr + (size_t)(b * N_ + j) * HFOUT_;
  __shared__ float red[256];
  for (int h = 0; h < H_; ++h) {
    float s = 0.0f;
    for (int f = threadIdx.x; f < FOUT_; f += 256) {
      float p = glp[h * FOUT_ + f] + grp[h * FOUT_ + f];
      p = p > 0.0f ? p : 0.2f * p;
      s += p * att[h * FOUT_ + f];
    }
    red[threadIdx.x] = s; __syncthreads();
    for (int off = 128; off > 0; off >>= 1) {
      if (threadIdx.x < off) red[threadIdx.x] += red[threadIdx.x + off];
      __syncthreads();
    }
    if (threadIdx.x == 0) scores[(size_t)bij * H_ + h] = red[0];
    __syncthreads();
  }
}

// ---------------- masked softmax + aggregation + fused fc5 partial ----------------
__global__ __launch_bounds__(256) void gat_out_kernel(
    const float* __restrict__ gl, const float* __restrict__ scores,
    const float* __restrict__ adj, const float* __restrict__ gat_bias,
    const float* __restrict__ fc5w, float* __restrict__ prepart) {
  int id = blockIdx.x;
  int h = id & 3, i = (id >> 2) % N_, b = id / (H_ * N_);
  __shared__ float alpha[N_];
  if (threadIdx.x < N_) {
    int j = threadIdx.x;
    bool m = (adj[j * N_ + i] > 0.0f) || (i == j);  // mask = adj.T + I > 0
    float sc = scores[(((size_t)b * N_ + i) * N_ + j) * H_ + h];
    alpha[j] = m ? sc : -__builtin_inff();
  }
  __syncthreads();
  if (threadIdx.x == 0) {
    float mx = -__builtin_inff();
    for (int j = 0; j < N_; ++j) mx = fmaxf(mx, alpha[j]);
    float sum = 0.0f;
    for (int j = 0; j < N_; ++j) { float e = __expf(alpha[j] - mx); alpha[j] = e; sum += e; }
    float inv = 1.0f / sum;
    for (int j = 0; j < N_; ++j) alpha[j] *= inv;
  }
  __syncthreads();
  float part = 0.0f;
  for (int f = threadIdx.x; f < FOUT_; f += 256) {
    float acc = 0.0f;
    for (int j = 0; j < N_; ++j)
      acc += alpha[j] * gl[(size_t)(b * N_ + j) * HFOUT_ + h * FOUT_ + f];
    float v = acc + gat_bias[h * FOUT_ + f];
    v = v > 0.0f ? v : 0.0f;
    part += v * fc5w[(size_t)i * HFOUT_ + h * FOUT_ + f];
  }
  __shared__ float red[256];
  red[threadIdx.x] = part; __syncthreads();
  for (int off = 128; off > 0; off >>= 1) {
    if (threadIdx.x < off) red[threadIdx.x] += red[threadIdx.x + off];
    __syncthreads();
  }
  if (threadIdx.x == 0) prepart[(size_t)b * (N_ * H_) + i * H_ + h] = red[0];
}

__global__ __launch_bounds__(32) void finalize_pre_kernel(
    const float* __restrict__ prepart, const float* __restrict__ fc5b,
    float* __restrict__ pre) {
  int b = threadIdx.x;
  if (b < B_) {
    float s = fc5b[0];
    for (int k = 0; k < N_ * H_; ++k) s += prepart[(size_t)b * (N_ * H_) + k];
    pre[b] = s;
  }
}

// ---------------- host orchestration ----------------
extern "C" void kernel_launch(void* const* d_in, const int* in_sizes, int n_in,
                              void* d_out, int out_size, void* d_ws, size_t ws_size,
                              hipStream_t stream) {
  (void)in_sizes; (void)n_in; (void)out_size; (void)ws_size;
  const float* data    = (const float*)d_in[0];
  const float* pe      = (const float*)d_in[1];
  const float* w_start = (const float*)d_in[2];
  const float* b_start = (const float*)d_in[3];
  const float* dw[3][3];
  const float* db[3][3];
  for (int br = 0; br < 3; ++br)
    for (int l = 0; l < 3; ++l) {
      dw[br][l] = (const float*)d_in[4 + br * 3 + l];
      db[br][l] = (const float*)d_in[13 + br * 3 + l];
    }
  const float* fc1w    = (const float*)d_in[22];
  const float* fc1b    = (const float*)d_in[23];
  const float* wl      = (const float*)d_in[24];
  const float* bl      = (const float*)d_in[25];
  const float* wr      = (const float*)d_in[26];
  const float* brr     = (const float*)d_in[27];
  const float* att     = (const float*)d_in[28];
  const float* gatb    = (const float*)d_in[29];
  const float* fc5w    = (const float*)d_in[30];
  const float* fc5b    = (const float*)d_in[31];

  float* out  = (float*)d_out;
  float* pre  = out;                        // [0, 32)
  float* pel  = out + 32;                   // [32, 32+9216)
  float* adjo = out + 32 + N_ * B_ * P_;    // [9248, 9572)

  // workspace layout (256B aligned)
  char* ws = (char*)d_ws;
  size_t off = 0;
  auto alloc = [&](size_t bytes) -> char* {
    char* p = ws + off;
    off = (off + bytes + 255) & ~(size_t)255;
    return p;
  };
  float* xin    = (float*)alloc((size_t)B_ * CIN_  * N_ * TP_ * 4);
  float* t0     = (float*)alloc((size_t)B_ * COUT_ * N_ * TP_ * 4);
  float* t1     = (float*)alloc((size_t)B_ * COUT_ * N_ * TP_ * 4);
  float* Fb     = (float*)alloc((size_t)B_ * COUT_ * N_ * TP_ * 4);
  float* Gb     = (float*)alloc((size_t)B_ * COUT_ * N_ * TP_ * 4);
  float* Rb     = (float*)alloc((size_t)B_ * COUT_ * N_ * TP_ * 4);
  __bf16* Ahi   = (__bf16*)alloc((size_t)MROWS_ * FIN_ * 2);
  __bf16* Alo   = (__bf16*)alloc((size_t)MROWS_ * FIN_ * 2);
  float* gl     = (float*)alloc((size_t)MROWS_ * HFOUT_ * 4);
  float* gr     = (float*)alloc((size_t)MROWS_ * HFOUT_ * 4);
  float* scores = (float*)alloc((size_t)B_ * N_ * N_ * H_ * 4);
  float* Smat   = (float*)alloc((size_t)N_ * N_ * 4);
  float* adjw   = (float*)alloc((size_t)N_ * N_ * 4);
  float* prep   = (float*)alloc((size_t)B_ * N_ * H_ * 4);

  const int BN = B_ * N_;

  // 1) start embed + pad
  start_kernel<<<BN, 256, 0, stream>>>(data, w_start, b_start, xin);

  // 2) three DCN branches (dilations 1,2,4)
  float* branch_out[3] = {Fb, Gb, Rb};
  for (int br = 0; br < 3; ++br) {
    conv_kernel<<<BN, 256, 0, stream>>>(xin, dw[br][0], db[br][0], t0, CIN_, 1);
    conv_kernel<<<BN, 256, 0, stream>>>(t0,  dw[br][1], db[br][1], t1, COUT_, 2);
    conv_kernel<<<BN, 256, 0, stream>>>(t1,  dw[br][2], db[br][2], branch_out[br], COUT_, 4);
  }

  // 3) GLU gate -> split-bf16 A matrix + pel output
  gate_kernel<<<BN, 256, 0, stream>>>(Fb, Gb, Rb, fc1w, fc1b, Ahi, Alo, pel);

  // 4) Kendall adjacency
  kendall_S_kernel<<<N_ * N_, 256, 0, stream>>>(pe, Smat);
  kendall_adj_kernel<<<1, 384, 0, stream>>>(Smat, adjw, adjo);

  // 5) the two big projections on the WMMA path
  const int gemm_blocks = (MROWS_ / 16) * (HFOUT_ / 16) / 8; // 1728, exact
  gemm_wmma_kernel<<<gemm_blocks, 256, 0, stream>>>(Ahi, Alo, wl, bl, gl);
  gemm_wmma_kernel<<<gemm_blocks, 256, 0, stream>>>(Ahi, Alo, wr, brr, gr);

  // 6) attention scores
  scores_kernel<<<B_ * N_ * N_, 256, 0, stream>>>(gl, gr, att, scores);

  // 7) masked softmax + aggregation + fused fc5 partials
  gat_out_kernel<<<B_ * N_ * H_, 256, 0, stream>>>(gl, scores, adjw, gatb, fc5w, prep);

  // 8) deterministic final reduction into pre
  finalize_pre_kernel<<<1, 32, 0, stream>>>(prep, fc5b, pre);
}
